// GAT_18004502905469
// MI455X (gfx1250) — matrix operand order repaired and checked by reading, not compiled, not run on hardware
//
#include <hip/hip_runtime.h>
#include <hip/hip_bf16.h>
#include <math.h>

#define BS 8
#define NN 2000
#define NPAD 2016   // 63 * 32
#define NMAIN 1984  // 62 branch-free chunks; tail chunk handles [1984,2016)

typedef __attribute__((ext_vector_type(16))) __bf16 v16bf;
typedef __attribute__((ext_vector_type(8)))  float  v8f;

__device__ __forceinline__ unsigned short f2bf(float f) {
  unsigned int u = __float_as_uint(f);
  u += 0x7FFFu + ((u >> 16) & 1u);          // round-to-nearest-even
  return (unsigned short)(u >> 16);
}
__device__ __forceinline__ float elu_(float v)   { return v > 0.f ? v : expm1f(v); }
__device__ __forceinline__ float lrelu_(float v) { return fmaxf(v, 0.2f * v); }

// ---------------------------------------------------------------------------
// Layer 1: rank-1 attention, one block per (row i, batch b).
// out: h1[b,n,32], z[b,0:N] = x0
// ---------------------------------------------------------------------------
__global__ __launch_bounds__(256)
void gat_layer1(const float* __restrict__ x, const float* __restrict__ adj,
                const float* __restrict__ W1, const float* __restrict__ a1,
                float* __restrict__ h1, float* __restrict__ z)
{
  __shared__ float s_adj[NN];
  __shared__ float s_x[NN];
  __shared__ float red[256];
  const int i = blockIdx.x, b = blockIdx.y, tid = threadIdx.x;
  for (int t = tid; t < NN; t += 256) {
    s_adj[t] = adj[(size_t)i * NN + t];
    s_x[t]   = x[b * NN + t];
  }
  __syncthreads();
  const float xi = s_x[i];
  if (tid == 0) z[b * 3 * NN + i] = xi;   // x0 = mean over feature dim of size 1

  // row mean of x (exact fallback for fully-masked rows: softmax(all NEG) is uniform)
  float ms = 0.f;
  for (int j = tid; j < NN; j += 256) ms += s_x[j];
  red[tid] = ms; __syncthreads();
  for (int s2 = 128; s2 > 0; s2 >>= 1) { if (tid < s2) red[tid] += red[tid + s2]; __syncthreads(); }
  const float meanx = red[0] * (1.f / NN);
  __syncthreads();

  for (int h = 0; h < 4; ++h) {
    float cs = 0.f, cd = 0.f;
    for (int f = 0; f < 8; ++f) {
      const float w = W1[h * 8 + f];
      cs += w * a1[h * 16 + f];
      cd += w * a1[h * 16 + 8 + f];
    }
    const float li = cs * xi;
    // pass 1: masked row max
    float mx = -INFINITY;
    for (int j = tid; j < NN; j += 256)
      if (s_adj[j] > 0.f) mx = fmaxf(mx, lrelu_(li + cd * s_x[j]));
    red[tid] = mx; __syncthreads();
    for (int s2 = 128; s2 > 0; s2 >>= 1) { if (tid < s2) red[tid] = fmaxf(red[tid], red[tid + s2]); __syncthreads(); }
    const float M = red[0]; __syncthreads();
    // pass 2: exp sum and weighted sum (mask-by-multiply, adj is {0,1})
    float sp = 0.f, tp = 0.f;
    if (M > -INFINITY) {
      for (int j = tid; j < NN; j += 256) {
        const float p = s_adj[j] * __expf(lrelu_(li + cd * s_x[j]) - M);
        sp += p; tp += p * s_x[j];
      }
    }
    red[tid] = sp; __syncthreads();
    for (int s2 = 128; s2 > 0; s2 >>= 1) { if (tid < s2) red[tid] += red[tid + s2]; __syncthreads(); }
    const float S = red[0]; __syncthreads();
    red[tid] = tp; __syncthreads();
    for (int s2 = 128; s2 > 0; s2 >>= 1) { if (tid < s2) red[tid] += red[tid + s2]; __syncthreads(); }
    const float T = red[0]; __syncthreads();
    const float tn = (S > 0.f) ? (T / S) : meanx;
    if (tid < 8)
      h1[((size_t)(b * NN) + i) * 32 + h * 8 + tid] = elu_(tn * W1[h * 8 + tid]);
  }
}

// ---------------------------------------------------------------------------
// Layer 2 projections: h2pre = h1 @ W2[head]; emit bf16 transposed [bh][f][npad]
// (WMMA B operand), plus src2/dst2 logit halves.
// ---------------------------------------------------------------------------
__global__ __launch_bounds__(256)
void gat_layer2_proj(const float* __restrict__ h1, const float* __restrict__ W2,
                     const float* __restrict__ a2,
                     unsigned short* __restrict__ h2bT,
                     float* __restrict__ src2, float* __restrict__ dst2)
{
  __shared__ float W2s[3 * 32 * 16];
  __shared__ float a2s[3 * 32];
  const int tid = threadIdx.x;
  for (int t = tid; t < 3 * 32 * 16; t += 256) W2s[t] = W2[t];
  for (int t = tid; t < 3 * 32;      t += 256) a2s[t] = a2[t];
  __syncthreads();
  const int idx = blockIdx.x * 256 + tid;
  if (idx >= BS * NN) return;
  const int b = idx / NN, n = idx % NN;
  float hv[32];
  for (int c = 0; c < 32; ++c) hv[c] = h1[((size_t)(b * NN) + n) * 32 + c];
  for (int hd = 0; hd < 3; ++hd) {
    float hp[16];
    for (int f = 0; f < 16; ++f) {
      float acc = 0.f;
      for (int c = 0; c < 32; ++c) acc += hv[c] * W2s[(hd * 32 + c) * 16 + f];
      hp[f] = acc;
    }
    float sr = 0.f, ds = 0.f;
    for (int f = 0; f < 16; ++f) { sr += hp[f] * a2s[hd * 32 + f]; ds += hp[f] * a2s[hd * 32 + 16 + f]; }
    const int bh = b * 3 + hd;
    src2[bh * NN + n] = sr;
    dst2[bh * NN + n] = ds;
    for (int f = 0; f < 16; ++f)
      h2bT[((size_t)(bh * 16) + f) * NPAD + n] = f2bf(hp[f]);
  }
}

__global__ void pad_h2bT(unsigned short* __restrict__ h2bT)
{
  const int idx = blockIdx.x * 256 + threadIdx.x;
  const int PAD = NPAD - NN;                      // 16
  if (idx >= BS * 3 * 16 * PAD) return;
  const int row = idx / PAD;
  const int n   = NN + idx % PAD;
  h2bT[(size_t)row * NPAD + n] = 0;
}

// ---------------------------------------------------------------------------
// Layer 2 attention aggregation with V_WMMA_F32_16X16X32_BF16.
// One wave per 16-row tile of one (b,head). Single pass: unnormalized
// probabilities (masked -> exactly 0), WMMA-accumulate att*h and VALU row
// sums, divide + ELU at the end. Main loop branch-free; tail chunk guarded.
// All chunk data (adj tile, dst2 runs, B tile) comes from plain vector
// global loads -> one clause-groupable vmem batch per chunk, no LDS waits.
// ---------------------------------------------------------------------------
template<bool GUARD>
__device__ __forceinline__ void attn_chunk(int j0, int ka, int kb, int half,
                                           float si,
                                           const float* __restrict__ adjrow,
                                           const float* __restrict__ drow,
                                           const unsigned short* __restrict__ hb,
                                           v8f& acc, float* __restrict__ rs4)
{
  const int c0 = j0 + ka, c1 = j0 + kb;           // multiples of 8 -> float4 aligned

  // adjacency tile rows (this lane's 16 K slots)
  float av[16];
  // dst2 values for the same K slots (same addresses across each half-wave)
  float dv[16];
  {
    float4 q0 = make_float4(0.f, 0.f, 0.f, 0.f), q1 = q0, q2 = q0, q3 = q0;
    float4 e0 = q0, e1 = q0, e2 = q0, e3 = q0;
    if (!GUARD || c0 < NN) {
      q0 = *(const float4*)(adjrow + c0); q1 = *(const float4*)(adjrow + c0 + 4);
      e0 = *(const float4*)(drow   + c0); e1 = *(const float4*)(drow   + c0 + 4);
    }
    if (!GUARD || c1 < NN) {
      q2 = *(const float4*)(adjrow + c1); q3 = *(const float4*)(adjrow + c1 + 4);
      e2 = *(const float4*)(drow   + c1); e3 = *(const float4*)(drow   + c1 + 4);
    }
    av[0]=q0.x; av[1]=q0.y; av[2]=q0.z;  av[3]=q0.w;
    av[4]=q1.x; av[5]=q1.y; av[6]=q1.z;  av[7]=q1.w;
    av[8]=q2.x; av[9]=q2.y; av[10]=q2.z; av[11]=q2.w;
    av[12]=q3.x;av[13]=q3.y;av[14]=q3.z; av[15]=q3.w;
    dv[0]=e0.x; dv[1]=e0.y; dv[2]=e0.z;  dv[3]=e0.w;
    dv[4]=e1.x; dv[5]=e1.y; dv[6]=e1.z;  dv[7]=e1.w;
    dv[8]=e2.x; dv[9]=e2.y; dv[10]=e2.z; dv[11]=e2.w;
    dv[12]=e3.x;dv[13]=e3.y;dv[14]=e3.z; dv[15]=e3.w;
  }

  // unnormalized masked probabilities (adj is exactly {0,1} -> multiply mask);
  // 4-way split row-sum accumulation to keep dependency depth shallow.
  float pf[16];
  #pragma unroll
  for (int u = 0; u < 16; ++u) {
    const float l = si + dv[u];
    const float p = av[u] * __expf(fmaxf(l, 0.2f * l));
    rs4[u & 3] += p;
    pf[u] = p;
  }

  // A operand: element u <-> this lane's u-th K slot (native v_cvt_pk_bf16_f32)
  v16bf Av;
  #pragma unroll
  for (int u = 0; u < 16; ++u) Av[u] = (__bf16)pf[u];

  // B operand: K-pair dword loads from transposed bf16 h2 projection
  union { v16bf v; unsigned int u[8]; } B;
  const int nb = j0 + (half ? 8 : 0);
  #pragma unroll
  for (int v = 0; v < 8; ++v)
    B.u[v] = *(const unsigned int*)(hb + nb + 2 * v + (v >= 4 ? 8 : 0));

  acc = __builtin_amdgcn_wmma_f32_16x16x32_bf16(false, Av, false, B.v,
                                                (short)0, acc, false, false);
}

__global__ __launch_bounds__(256)
void gat_layer2_attn(const float* __restrict__ adj,
                     const float* __restrict__ src2, const float* __restrict__ dst2,
                     const unsigned short* __restrict__ h2bT,
                     float* __restrict__ h2)
{
  const int lane = threadIdx.x & 31;
  const int wv   = threadIdx.x >> 5;
  const int t    = blockIdx.x * 8 + wv;           // 3000 tiles = 375 blocks * 8 waves (exact)
  if (t >= BS * 3 * 125) return;
  const int b    = t / 375;
  const int r    = t % 375;
  const int head = r / 125;
  const int i0   = (r % 125) * 16;
  const int m    = lane & 15;                     // row (A) / column f (B,D)
  const int half = lane >> 4;
  const int bh   = b * 3 + head;
  const float  si      = src2[bh * NN + i0 + m];
  const float* adjrow  = adj + (size_t)(i0 + m) * NN;
  const float* drow    = dst2 + bh * NN;
  const unsigned short* hb = h2bT + ((size_t)(bh * 16) + m) * NPAD;
  const int ka = half ? 8 : 0;                    // A-layout K bases per lane half
  const int kb = ka + 16;

  v8f acc = {0.f, 0.f, 0.f, 0.f, 0.f, 0.f, 0.f, 0.f};
  float rs4[4] = {0.f, 0.f, 0.f, 0.f};

  #pragma unroll 2
  for (int j0 = 0; j0 < NMAIN; j0 += 32)
    attn_chunk<false>(j0, ka, kb, half, si, adjrow, drow, hb, acc, rs4);
  attn_chunk<true>(NMAIN, ka, kb, half, si, adjrow, drow, hb, acc, rs4);

  const float rs  = (rs4[0] + rs4[1]) + (rs4[2] + rs4[3]);
  const float rsf = rs + __shfl_xor(rs, 16, 32);  // full row sum, keyed by lane%16
  #pragma unroll
  for (int rr = 0; rr < 8; ++rr) {
    const int   M     = half ? (8 + rr) : rr;     // C/D layout row for VGPR rr
    const float denom = __shfl(rsf, M, 32);
    const float val   = elu_(acc[rr] / denom);
    h2[((size_t)(b * NN) + i0 + M) * 48 + head * 16 + m] = val;
  }
}

// ---------------------------------------------------------------------------
// z = concat([x0, x1, x2]); x1/x2 are per-node dots with pw1/pw2.
// ---------------------------------------------------------------------------
__global__ __launch_bounds__(256)
void assemble_z(const float* __restrict__ h1, const float* __restrict__ h2,
                const float* __restrict__ pw1, const float* __restrict__ pb1,
                const float* __restrict__ pw2, const float* __restrict__ pb2,
                float* __restrict__ z)
{
  const int idx = blockIdx.x * 256 + threadIdx.x;
  if (idx >= BS * NN) return;
  const int b = idx / NN, n = idx % NN;
  float s1 = pb1[0];
  const float* r1 = h1 + ((size_t)(b * NN) + n) * 32;
  for (int c = 0; c < 32; ++c) s1 += r1[c] * pw1[c];
  z[b * 3 * NN + NN + n] = s1;
  float s2 = pb2[0];
  const float* r2 = h2 + ((size_t)(b * NN) + n) * 48;
  for (int c = 0; c < 48; ++c) s2 += r2[c] * pw2[c];
  z[b * 3 * NN + 2 * NN + n] = s2;
}

// ---------------------------------------------------------------------------
// MLP head. fc1 uses deterministic split-K (separate partial buffers, no
// float atomics -> bitwise-identical across graph replays).
// ---------------------------------------------------------------------------
__global__ __launch_bounds__(128)
void fc_split(const float* __restrict__ in, const float* __restrict__ w,
              float* __restrict__ part, int K, int C)
{
  const int c = blockIdx.x * 128 + threadIdx.x;
  if (c >= C) return;
  const int ns = gridDim.y, s = blockIdx.y;
  const int kc = K / ns;
  const int k0 = s * kc, k1 = k0 + kc;
  float acc[BS];
  for (int b = 0; b < BS; ++b) acc[b] = 0.f;
  for (int k = k0; k < k1; ++k) {
    const float wv = w[(size_t)k * C + c];
    for (int b = 0; b < BS; ++b) acc[b] += in[b * K + k] * wv;
  }
  float* po = part + (size_t)s * BS * C;
  for (int b = 0; b < BS; ++b) po[b * C + c] = acc[b];
}

__global__ __launch_bounds__(128)
void fc_reduce(const float* __restrict__ part, const float* __restrict__ bias,
               float* __restrict__ out, int C, int ns, int act)
{
  const int idx = blockIdx.x * 128 + threadIdx.x;
  if (idx >= BS * C) return;
  float s = bias[idx % C];
  for (int p = 0; p < ns; ++p) s += part[(size_t)p * BS * C + idx];
  out[idx] = act ? elu_(s) : s;
}

__global__ __launch_bounds__(128)
void fc_direct(const float* __restrict__ in, const float* __restrict__ w,
               const float* __restrict__ bias, float* __restrict__ out,
               int K, int C, int act)
{
  const int c = blockIdx.x * 128 + threadIdx.x;
  if (c >= C) return;
  float acc[BS];
  for (int b = 0; b < BS; ++b) acc[b] = bias[c];
  for (int k = 0; k < K; ++k) {
    const float wv = w[(size_t)k * C + c];
    for (int b = 0; b < BS; ++b) acc[b] += in[b * K + k] * wv;
  }
  for (int b = 0; b < BS; ++b) out[b * C + c] = act ? elu_(acc[b]) : acc[b];
}

// ---------------------------------------------------------------------------
extern "C" void kernel_launch(void* const* d_in, const int* in_sizes, int n_in,
                              void* d_out, int out_size, void* d_ws, size_t ws_size,
                              hipStream_t stream)
{
  (void)in_sizes; (void)n_in; (void)out_size; (void)ws_size;
  const float* x     = (const float*)d_in[0];
  const float* adj   = (const float*)d_in[1];
  const float* W1    = (const float*)d_in[2];
  const float* a1    = (const float*)d_in[3];
  const float* W2    = (const float*)d_in[4];
  const float* a2    = (const float*)d_in[5];
  const float* pw1   = (const float*)d_in[6];
  const float* pb1   = (const float*)d_in[7];
  const float* pw2   = (const float*)d_in[8];
  const float* pb2   = (const float*)d_in[9];
  const float* fc1_w = (const float*)d_in[10];
  const float* fc1_b = (const float*)d_in[11];
  const float* fc2_w = (const float*)d_in[12];
  const float* fc2_b = (const float*)d_in[13];
  const float* fc3_w = (const float*)d_in[14];
  const float* fc3_b = (const float*)d_in[15];
  const float* fc4_w = (const float*)d_in[16];
  const float* fc4_b = (const float*)d_in[17];
  const float* fc5_w = (const float*)d_in[18];
  const float* fc5_b = (const float*)d_in[19];

  char* ws = (char*)d_ws;
  size_t off = 0;
  auto alloc = [&](size_t bytes) { size_t o = off; off += (bytes + 255) & ~(size_t)255; return o; };

  float*          h1    = (float*)(ws + alloc(sizeof(float) * BS * NN * 32));
  float*          h2    = (float*)(ws + alloc(sizeof(float) * BS * NN * 48));
  unsigned short* h2bT  = (unsigned short*)(ws + alloc(sizeof(unsigned short) * BS * 3 * 16 * NPAD));
  float*          src2  = (float*)(ws + alloc(sizeof(float) * BS * 3 * NN));
  float*          dst2  = (float*)(ws + alloc(sizeof(float) * BS * 3 * NN));
  float*          z     = (float*)(ws + alloc(sizeof(float) * BS * 3 * NN));
  float*          t1    = (float*)(ws + alloc(sizeof(float) * BS * 600));
  float*          t2    = (float*)(ws + alloc(sizeof(float) * BS * 256));
  float*          t4    = (float*)(ws + alloc(sizeof(float) * BS * 32));
  float*          part  = (float*)(ws + alloc(sizeof(float) * 10 * BS * 600));

  gat_layer1<<<dim3(NN, BS), 256, 0, stream>>>(x, adj, W1, a1, h1, z);
  gat_layer2_proj<<<(BS * NN + 255) / 256, 256, 0, stream>>>(h1, W2, a2, h2bT, src2, dst2);
  pad_h2bT<<<(BS * 3 * 16 * (NPAD - NN) + 255) / 256, 256, 0, stream>>>(h2bT);
  gat_layer2_attn<<<(BS * 3 * 125) / 8, 256, 0, stream>>>(adj, src2, dst2, h2bT, h2);
  assemble_z<<<(BS * NN + 255) / 256, 256, 0, stream>>>(h1, h2, pw1, pb1, pw2, pb2, z);

  // fc1: [8,6000] x [6000,600], split-K 10, deterministic reduce + ELU
  fc_split<<<dim3((600 + 127) / 128, 10), 128, 0, stream>>>(z, fc1_w, part, 3 * NN, 600);
  fc_reduce<<<(BS * 600 + 127) / 128, 128, 0, stream>>>(part, fc1_b, t1, 600, 10, 1);
  fc_direct<<<(256 + 127) / 128, 128, 0, stream>>>(t1, fc2_w, fc2_b, t2, 600, 256, 1);

  float* z1 = (float*)d_out;                       // [8,64] output 0
  fc_direct<<<1, 128, 0, stream>>>(t2, fc3_w, fc3_b, z1, 256, 64, 1);
  fc_direct<<<1, 128, 0, stream>>>(z1, fc4_w, fc4_b, t4, 64, 32, 1);
  fc_direct<<<1, 128, 0, stream>>>(t4, fc5_w, fc5_b, z1 + BS * 64, 32, 2, 0);  // gat_out [8,2]
}
